// MultiHeadAttention_65051574665284
// MI455X (gfx1250) — matrix operand order repaired and checked by reading, not compiled, not run on hardware
//
#include <hip/hip_runtime.h>
#include <hip/hip_bf16.h>

// ---------------------------------------------------------------------------
// MHA block (proj -> flash attention -> out-proj + residual -> LayerNorm)
// for MI455X / gfx1250. bf16 WMMA on all matmul paths, fp32 accumulation.
// Softmax row-sums computed on the matrix pipe (P x ones), row-max via
// v_permlane16 VALU butterflies -- no ds_bpermute on the hot path.
// ---------------------------------------------------------------------------

typedef __bf16 bf16;
typedef __attribute__((ext_vector_type(16))) __bf16 v16bf_t;
typedef __attribute__((ext_vector_type(8)))  __bf16 v8bf_t;
typedef __attribute__((ext_vector_type(8)))  float  v8f_t;

#define D_MODEL 1024
#define NHEADS  16
#define DH      64
#define SEQ     2048
#define BS      4
#define MTOT    (BS * SEQ)          // 8192 rows
#define LN_EPS  1e-5f

// log2(e), and attention scale folded into log2 domain
#define LOG2E      1.4426950408889634f
#define QSCALE     (0.125f * LOG2E)        // 1/sqrt(64) * log2(e), folded into Q
#define MASKSCALE  (-1.0e9f * LOG2E)       // mask coefficient in log2 domain

// ---------------- fragment loaders (per documented gfx1250 WMMA layouts) ----
__device__ __forceinline__ v16bf_t load_a16(const bf16* __restrict__ row, int k, int half) {
  const v8bf_t lo = *(const v8bf_t*)(row + k + half * 8);
  const v8bf_t hi = *(const v8bf_t*)(row + k + half * 8 + 16);
  v16bf_t f;
#pragma unroll
  for (int i = 0; i < 8; ++i) { f[i] = lo[i]; f[i + 8] = hi[i]; }
  return f;
}

__device__ __forceinline__ v16bf_t load_b16(const bf16* __restrict__ row, int k, int half) {
  const v8bf_t lo = *(const v8bf_t*)(row + k + half * 8);
  const v8bf_t hi = *(const v8bf_t*)(row + k + half * 8 + 16);
  v16bf_t f;
#pragma unroll
  for (int i = 0; i < 8; ++i) { f[i] = lo[i]; f[i + 8] = hi[i]; }
  return f;
}

__device__ __forceinline__ v8f_t wmma_bf16(v16bf_t a, v16bf_t b, v8f_t c) {
  return __builtin_amdgcn_wmma_f32_16x16x32_bf16(false, a, false, b, (short)0, c,
                                                 false, false);
}

// xor-butterfly max over the 16 lanes of each half-wave, pure VALU
// (v_permlane16_b32: nibble i of {selHi,selLo} = source lane for lane i%16)
__device__ __forceinline__ float xor16_max(float v) {
  int x = __float_as_int(v);
  int t;
  t = __builtin_amdgcn_permlane16(x, x, 0x67452301u, 0xEFCDAB89u, false, false); // ^1
  v = fmaxf(v, __int_as_float(t)); x = __float_as_int(v);
  t = __builtin_amdgcn_permlane16(x, x, 0x54761032u, 0xDCFE98BAu, false, false); // ^2
  v = fmaxf(v, __int_as_float(t)); x = __float_as_int(v);
  t = __builtin_amdgcn_permlane16(x, x, 0x32107654u, 0xBA98FEDCu, false, false); // ^4
  v = fmaxf(v, __int_as_float(t)); x = __float_as_int(v);
  t = __builtin_amdgcn_permlane16(x, x, 0xFEDCBA98u, 0x76543210u, false, false); // ^8
  v = fmaxf(v, __int_as_float(t));
  return v;
}

// ---------------- fp32 -> bf16 cast (8 elems / thread, one b128 store) ------
__global__ void cast_bf16_kernel(const float* __restrict__ in,
                                 bf16* __restrict__ out, int n8) {
  int i = blockIdx.x * blockDim.x + threadIdx.x;
  if (i < n8) {
    const float4 f0 = ((const float4*)in)[i * 2 + 0];
    const float4 f1 = ((const float4*)in)[i * 2 + 1];
    v8bf_t pk;
    pk[0] = (bf16)f0.x; pk[1] = (bf16)f0.y; pk[2] = (bf16)f0.z; pk[3] = (bf16)f0.w;
    pk[4] = (bf16)f1.x; pk[5] = (bf16)f1.y; pk[6] = (bf16)f1.z; pk[7] = (bf16)f1.w;
    *(v8bf_t*)(out + i * 8) = pk;
  }
}

// ---------------- projection GEMM: C = oscale * (X @ W^T), 64x64/wave -------
// MODE 0: store bf16, head-split layout   [b, h, s, dh]       (Q, K)
// MODE 1: store bf16, head-split T layout [b, h, dh, s]       (V, packed b128)
// MODE 2: store f32   [m, n] = acc + residual[m, n]           (Wo + residual)
template <int MODE>
__global__ __launch_bounds__(256) void gemm64_wmma(
    const bf16* __restrict__ X, const bf16* __restrict__ W,
    bf16* __restrict__ outB, float* __restrict__ outF,
    const float* __restrict__ residual, float oscale) {
  const int lane = threadIdx.x & 31;
  const int wave = threadIdx.x >> 5;
  const int half = lane >> 4;
  const int r    = lane & 15;
  const int tile = blockIdx.x * 8 + wave;       // 8 waves / block
  const int tilesN = D_MODEL / 64;              // 16
  const int tm = tile / tilesN;
  const int tn = tile % tilesN;

  const bf16* Arow[4];
  const bf16* Brow[4];
#pragma unroll
  for (int i = 0; i < 4; ++i) {
    Arow[i] = X + (size_t)(tm * 64 + i * 16 + r) * D_MODEL;
    Brow[i] = W + (size_t)(tn * 64 + i * 16 + r) * D_MODEL;
  }

  v8f_t acc[4][4];
#pragma unroll
  for (int mi = 0; mi < 4; ++mi)
#pragma unroll
    for (int ni = 0; ni < 4; ++ni) acc[mi][ni] = (v8f_t){};

#pragma unroll 1
  for (int k = 0; k < D_MODEL; k += 32) {
    v16bf_t a[4], b[4];
#pragma unroll
    for (int i = 0; i < 4; ++i) a[i] = load_a16(Arow[i], k, half);
#pragma unroll
    for (int i = 0; i < 4; ++i) b[i] = load_b16(Brow[i], k, half);
#pragma unroll
    for (int mi = 0; mi < 4; ++mi)
#pragma unroll
      for (int ni = 0; ni < 4; ++ni)
        acc[mi][ni] = wmma_bf16(a[mi], b[ni], acc[mi][ni]);
  }

#pragma unroll
  for (int mi = 0; mi < 4; ++mi) {
#pragma unroll
    for (int ni = 0; ni < 4; ++ni) {
      if (MODE == 1) {
        const int n  = tn * 64 + ni * 16 + r;
        const int m0 = tm * 64 + mi * 16 + 8 * half;   // first of 8 rows
        const int b_ = m0 >> 11;
        const int s  = m0 & (SEQ - 1);
        const int h_ = n >> 6, dh = n & (DH - 1);
        v8bf_t pk;
#pragma unroll
        for (int j = 0; j < 8; ++j) pk[j] = (bf16)acc[mi][ni][j];
        *(v8bf_t*)(outB + (((size_t)(b_ * NHEADS + h_)) * DH + dh) * SEQ + s) = pk;
      } else {
#pragma unroll
        for (int j = 0; j < 8; ++j) {
          const int m = tm * 64 + mi * 16 + j + 8 * half;
          const int n = tn * 64 + ni * 16 + r;
          if (MODE == 2) {
            const size_t idx = (size_t)m * D_MODEL + n;
            outF[idx] = acc[mi][ni][j] + residual[idx];
          } else {
            const int b_ = m >> 11;
            const int s  = m & (SEQ - 1);
            const int h_ = n >> 6, dh = n & (DH - 1);
            outB[(((size_t)(b_ * NHEADS + h_)) * SEQ + s) * DH + dh] =
                (bf16)(acc[mi][ni][j] * oscale);
          }
        }
      }
    }
  }
}

// ---------------- flash attention: one 16-row q-tile per wave, 64-key steps -
__global__ __launch_bounds__(128) void attn_fa16(
    const bf16* __restrict__ Qh, const bf16* __restrict__ Kh,
    const bf16* __restrict__ VhT, const float* __restrict__ zmask,
    bf16* __restrict__ ctx) {
  __shared__ __align__(16) bf16 psh[4][16][72];   // per-wave P staging (64+8 pad)

  const int lane = threadIdx.x & 31;
  const int wave = threadIdx.x >> 5;
  const int half = lane >> 4;
  const int r    = lane & 15;

  const int qblocks = SEQ / 64;                 // 32 q-blocks of 64 rows
  const int bh  = blockIdx.x / qblocks;         // b*H + h
  const int qb0 = (blockIdx.x % qblocks) * 64 + wave * 16;
  const int b_  = bh >> 4;
  const int h_  = bh & (NHEADS - 1);

  const bf16* Qp = Qh  + (size_t)bh * SEQ * DH;
  const bf16* Kp = Kh  + (size_t)bh * SEQ * DH;
  const bf16* Vt = VhT + (size_t)bh * DH * SEQ;
  const float* mb = zmask + (size_t)b_ * SEQ;

  const bf16* qrow = Qp + (size_t)(qb0 + r) * DH;
  const v16bf_t qa0 = load_a16(qrow, 0, half);   // Q (pre-scaled by QSCALE)
  const v16bf_t qa1 = load_a16(qrow, 32, half);

  v16bf_t onesb;
#pragma unroll
  for (int i = 0; i < 16; ++i) onesb[i] = (bf16)1.0f;

  float mrow[8];
  v8f_t Lacc = {};
  v8f_t O0 = {}, O1 = {}, O2 = {}, O3 = {};
#pragma unroll
  for (int j = 0; j < 8; ++j) mrow[j] = -3.0e38f;

#pragma unroll 1
  for (int kk = 0; kk < SEQ; kk += 64) {
    // ---- S = Q K^T (log2-domain: Q carries 0.125*log2e), 64 keys ----
    v8f_t S[4];
    float msk[4];
#pragma unroll
    for (int t = 0; t < 4; ++t) {
      const bf16* kr = Kp + (size_t)(kk + t * 16 + r) * DH;
      v8f_t s0 = {};
      s0 = wmma_bf16(qa0, load_b16(kr, 0, half), s0);
      s0 = wmma_bf16(qa1, load_b16(kr, 32, half), s0);
      S[t] = s0;
      msk[t] = mb[kk + t * 16 + r] * MASKSCALE;
    }

    // ---- online softmax (base-2): VALU-only row max, exp2, stage P in LDS --
#pragma unroll
    for (int j = 0; j < 8; ++j) {
      float s0 = S[0][j] + msk[0];
      float s1 = S[1][j] + msk[1];
      float s2 = S[2][j] + msk[2];
      float s3 = S[3][j] + msk[3];
      float mx = fmaxf(fmaxf(s0, s1), fmaxf(s2, s3));
      mx = xor16_max(mx);
      const float mnew  = fmaxf(mrow[j], mx);
      const float alpha = exp2f(mrow[j] - mnew);
      mrow[j] = mnew;
      const float p0 = exp2f(s0 - mnew);
      const float p1 = exp2f(s1 - mnew);
      const float p2 = exp2f(s2 - mnew);
      const float p3 = exp2f(s3 - mnew);
      Lacc[j] *= alpha;
      O0[j] *= alpha; O1[j] *= alpha; O2[j] *= alpha; O3[j] *= alpha;
      const int m = j + 8 * half;
      psh[wave][m][r]      = (bf16)p0;
      psh[wave][m][16 + r] = (bf16)p1;
      psh[wave][m][32 + r] = (bf16)p2;
      psh[wave][m][48 + r] = (bf16)p3;
    }
    asm volatile("s_wait_dscnt 0x0" ::: "memory");   // wave-local LDS visibility

    // ---- P as A-fragments; row sums on the matrix pipe (P x ones) ----
    const v16bf_t pa0 = load_a16(&psh[wave][r][0], 0, half);
    const v16bf_t pa1 = load_a16(&psh[wave][r][0], 32, half);
    Lacc = wmma_bf16(pa0, onesb, Lacc);
    Lacc = wmma_bf16(pa1, onesb, Lacc);

    // ---- O += P (16x64) x V (64x64), B = rows of V^T ----
    const bf16* v0 = Vt + (size_t)(0 * 16 + r) * SEQ;
    const bf16* v1 = Vt + (size_t)(1 * 16 + r) * SEQ;
    const bf16* v2 = Vt + (size_t)(2 * 16 + r) * SEQ;
    const bf16* v3 = Vt + (size_t)(3 * 16 + r) * SEQ;
    O0 = wmma_bf16(pa0, load_b16(v0, kk, half), O0);
    O1 = wmma_bf16(pa0, load_b16(v1, kk, half), O1);
    O2 = wmma_bf16(pa0, load_b16(v2, kk, half), O2);
    O3 = wmma_bf16(pa0, load_b16(v3, kk, half), O3);
    O0 = wmma_bf16(pa1, load_b16(v0, kk + 32, half), O0);
    O1 = wmma_bf16(pa1, load_b16(v1, kk + 32, half), O1);
    O2 = wmma_bf16(pa1, load_b16(v2, kk + 32, half), O2);
    O3 = wmma_bf16(pa1, load_b16(v3, kk + 32, half), O3);
  }

  // ---- normalize and write ctx in (b, s, h*64+dh) layout ----
#pragma unroll
  for (int j = 0; j < 8; ++j) {
    const float inv = 1.0f / Lacc[j];
    const int srow = qb0 + j + 8 * half;
    bf16* crow = ctx + ((size_t)(b_ * SEQ + srow)) * D_MODEL + h_ * DH;
    crow[0 * 16 + r] = (bf16)(O0[j] * inv);
    crow[1 * 16 + r] = (bf16)(O1[j] * inv);
    crow[2 * 16 + r] = (bf16)(O2[j] * inv);
    crow[3 * 16 + r] = (bf16)(O3[j] * inv);
  }
}

// ---------------- LayerNorm over last dim (1024), one row per block ---------
__global__ __launch_bounds__(256) void layernorm_kernel(
    const float* __restrict__ x, const float* __restrict__ gamma,
    const float* __restrict__ beta, float* __restrict__ out) {
  const int row = blockIdx.x;
  const float* xr = x + (size_t)row * D_MODEL;
  float s = 0.f, s2 = 0.f;
  for (int i = threadIdx.x; i < D_MODEL; i += 256) {
    const float v = xr[i];
    s += v; s2 += v * v;
  }
#pragma unroll
  for (int o = 16; o > 0; o >>= 1) {
    s  += __shfl_down(s, o);
    s2 += __shfl_down(s2, o);
  }
  __shared__ float red[16];
  __shared__ float mu_s, rstd_s;
  const int wv = threadIdx.x >> 5;
  if ((threadIdx.x & 31) == 0) { red[wv] = s; red[8 + wv] = s2; }
  __syncthreads();
  if (threadIdx.x == 0) {
    float a = 0.f, b = 0.f;
#pragma unroll
    for (int w = 0; w < 8; ++w) { a += red[w]; b += red[8 + w]; }
    const float mu  = a * (1.0f / D_MODEL);
    const float var = b * (1.0f / D_MODEL) - mu * mu;
    mu_s = mu;
    rstd_s = rsqrtf(var + LN_EPS);
  }
  __syncthreads();
  const float mu = mu_s, rstd = rstd_s;
  for (int i = threadIdx.x; i < D_MODEL; i += 256)
    out[(size_t)row * D_MODEL + i] = (xr[i] - mu) * rstd * gamma[i] + beta[i];
}

// ---------------------------------------------------------------------------
extern "C" void kernel_launch(void* const* d_in, const int* in_sizes, int n_in,
                              void* d_out, int out_size, void* d_ws, size_t ws_size,
                              hipStream_t stream) {
  const float* qf    = (const float*)d_in[0];
  const float* kf    = (const float*)d_in[1];
  const float* vf    = (const float*)d_in[2];
  const float* zmask = (const float*)d_in[3];
  const float* Wq    = (const float*)d_in[4];
  const float* Wk    = (const float*)d_in[5];
  const float* Wv    = (const float*)d_in[6];
  const float* Wo    = (const float*)d_in[7];
  const float* gamma = (const float*)d_in[8];
  const float* beta  = (const float*)d_in[9];
  float* outp = (float*)d_out;

  const size_t SZ_ACT = (size_t)MTOT * D_MODEL;     // 8M elems
  const size_t SZ_W   = (size_t)D_MODEL * D_MODEL;  // 1M elems

  bf16* qb  = (bf16*)d_ws;
  bf16* kb  = qb  + SZ_ACT;
  bf16* vb  = kb  + SZ_ACT;
  bf16* wqb = vb  + SZ_ACT;
  bf16* wkb = wqb + SZ_W;
  bf16* wvb = wkb + SZ_W;
  bf16* wob = wvb + SZ_W;
  bf16* Qh  = wob + SZ_W;
  bf16* Kh  = Qh  + SZ_ACT;
  bf16* VhT = Kh  + SZ_ACT;
  bf16* ctx = VhT + SZ_ACT;
  float* ao = (float*)(ctx + SZ_ACT);               // fp32, before LayerNorm

  // 1) casts to bf16
  {
    const int actN8 = (int)(SZ_ACT / 8), wN8 = (int)(SZ_W / 8);
    dim3 blk(256);
    cast_bf16_kernel<<<(actN8 + 255) / 256, blk, 0, stream>>>(qf, qb, actN8);
    cast_bf16_kernel<<<(actN8 + 255) / 256, blk, 0, stream>>>(kf, kb, actN8);
    cast_bf16_kernel<<<(actN8 + 255) / 256, blk, 0, stream>>>(vf, vb, actN8);
    cast_bf16_kernel<<<(wN8 + 255) / 256, blk, 0, stream>>>(Wq, wqb, wN8);
    cast_bf16_kernel<<<(wN8 + 255) / 256, blk, 0, stream>>>(Wk, wkb, wN8);
    cast_bf16_kernel<<<(wN8 + 255) / 256, blk, 0, stream>>>(Wv, wvb, wN8);
    cast_bf16_kernel<<<(wN8 + 255) / 256, blk, 0, stream>>>(Wo, wob, wN8);
  }

  // 2) Q/K/V projections: 64x64 tiles -> (128*16)/8 waves = 256 blocks
  //    Q carries the softmax scale (1/8 * log2e) folded in.
  {
    dim3 grid((MTOT / 64) * (D_MODEL / 64) / 8), blk(256);
    gemm64_wmma<0><<<grid, blk, 0, stream>>>(qb, wqb, Qh, nullptr, nullptr, QSCALE);
    gemm64_wmma<0><<<grid, blk, 0, stream>>>(kb, wkb, Kh, nullptr, nullptr, 1.0f);
    gemm64_wmma<1><<<grid, blk, 0, stream>>>(vb, wvb, VhT, nullptr, nullptr, 1.0f);
  }

  // 3) flash attention: (BS*H) * (SEQ/64) = 2048 blocks x 4 waves
  attn_fa16<<<dim3(BS * NHEADS * (SEQ / 64)), dim3(128), 0, stream>>>(
      Qh, Kh, VhT, zmask, ctx);

  // 4) output projection + residual (fp32 result)
  gemm64_wmma<2><<<dim3((MTOT / 64) * (D_MODEL / 64) / 8), dim3(256), 0, stream>>>(
      ctx, wob, nullptr, ao, qf, 1.0f);

  // 5) LayerNorm -> d_out
  layernorm_kernel<<<dim3(MTOT), dim3(256), 0, stream>>>(ao, gamma, beta, outp);
}
